// STFT_66116726555182
// MI455X (gfx1250) — compile-verified, matrix operand order checked by reading, not run on hardware
//
#include <hip/hip_runtime.h>
#include <hip/hip_bf16.h>
#include <math.h>

// ---------------------------------------------------------------------------
// STFT as WMMA GEMM for gfx1250 (MI455X, wave32).
//   mag/pha[b, bin, t],  bin in [0,257), t in [0,1999)
//   real[bin,t] = sum_n kern[bin      , n] * wav[b, t*256 + n]
//   imag[bin,t] = sum_n kern[bin + 512, n] * wav[b, t*256 + n]
// A = kernel rows (16 bins x 32 k, f16, pre-swizzled in d_ws),
// B = frame columns (32 k x 16 frames, f16 from LDS), C/D = f32 16x16.
// Each wave job: one A pair (real+imag) x four B frame-tiles -> 8 WMMA chains.
// ---------------------------------------------------------------------------

typedef __attribute__((ext_vector_type(16))) _Float16 v16h;
typedef __attribute__((ext_vector_type(2)))  __fp16   v2h;   // cvt_pkrtz result
typedef __attribute__((ext_vector_type(8)))  float    v8f;

#define NSAMP   512000      // samples per batch row
#define NBATCH  16
#define FLEN    512
#define HOP     256
#define NFRM    1999        // (512000-512)/256 + 1
#define NBIN    257         // 1024/4 + 1
#define KROW    512         // kernel row length (floats)
#define IMAG_OFF 512        // imag rows start at kern[512 + bin]
#define FT_PER_BLK 64       // frames per workgroup (four 16-frame B tiles)
#define NTT     4           // B tiles per job
#define SEG     (HOP*(FT_PER_BLK-1) + FLEN)   // 16640 samples staged
#define NBT     17          // bin tiles of 16 covering 0..271 (bin<257 masked)
#define NKC     16          // K chunks of 32 (512 total)
#define NROWT   64          // 1024 rows / 16
#define PADHOP  (HOP + 2)   // padded per-frame column stride in halves
#define EPS_F   1.1920928955078125e-07f

// Pre-swizzled A buffer: a16[tile][kk][lane][j], j in [0,16) halves.
//   row  = tile*16 + (lane&15)
//   koff = kk*32 + ((lane&16)?8:0);  k = koff + j (j<8), koff + 8 + j (j>=8)
#define A16_GROUPS (NROWT * NKC * 32)            // 32768 fragments
#define A16_BYTES  ((size_t)A16_GROUPS * 16 * 2) // 1 MiB

// LDS: SEG halves + 2-half pad every 256 halves. padded(h) = h + (h>>8)*2.
// Max padded index 16639 + 128 = 16767.
#define LDS_HALVES 16832

// Build one 16-half A fragment from f32 kernel rows (prepass + fallback).
__device__ __forceinline__ v16h cvt_a_frag(const float* __restrict__ src) {
    v16h a;
#pragma unroll
    for (int j = 0; j < 4; ++j) {
        v2h c0 = __builtin_amdgcn_cvt_pkrtz(src[2*j],      src[2*j+1]);
        v2h c1 = __builtin_amdgcn_cvt_pkrtz(src[16 + 2*j], src[16 + 2*j+1]);
        a[2*j]       = (_Float16)c0.x;  a[2*j+1]     = (_Float16)c0.y;
        a[8 + 2*j]   = (_Float16)c1.x;  a[8 + 2*j+1] = (_Float16)c1.y;
    }
    return a;
}

// ---- Prepass: f32 kernel -> f16 A fragments in WMMA lane order -------------
__global__ __launch_bounds__(256)
void swizzle_a16_kernel(const float* __restrict__ kern,
                        _Float16* __restrict__ a16) {
    const int g = blockIdx.x * blockDim.x + threadIdx.x;  // fragment id
    if (g >= A16_GROUPS) return;
    const int lane = g & 31;
    const int kk   = (g >> 5) & (NKC - 1);
    const int tile = g >> 9;
    const int row  = tile * 16 + (lane & 15);
    const int koff = kk * 32 + ((lane & 16) ? 8 : 0);
    const v16h frag = cvt_a_frag(kern + (size_t)row * KROW + koff);
    *(v16h*)(a16 + (size_t)g * 16) = frag;
}

// B fragment from LDS given a pre-padded half-index base of the first 8-half
// chunk; second chunk is +16 halves (never crosses a 256-half row).
__device__ __forceinline__ v16h load_b_raw(const _Float16* __restrict__ sw,
                                           int hbase) {
    v16h b;
#pragma unroll
    for (int j = 0; j < 8; ++j) {
        b[j]     = sw[hbase + j];
        b[8 + j] = sw[hbase + 16 + j];
    }
    return b;
}

// ---- Main GEMM kernel (PRE=1: pre-swizzled f16 A; PRE=0: cvt in loop) ------
template <bool PRE>
__global__ __launch_bounds__(256)
void stft_wmma_kernel(const float* __restrict__ wav,
                      const float* __restrict__ kern,
                      const _Float16* __restrict__ a16,
                      float* __restrict__ mag,
                      float* __restrict__ pha) {
    __shared__ __align__(16) _Float16 sw[LDS_HALVES];

    const int tid   = threadIdx.x;
    const int lane  = tid & 31;
    const int wave  = tid >> 5;
    const int ftile = blockIdx.x;           // frame tile [0,32)
    const int batch = blockIdx.y;           // [0,16)
    const int t0    = ftile * FT_PER_BLK;

    // ---- Stage wav segment into LDS as f16 (coalesced f32 loads + pk cvt) --
    {
        const float* wsrc = wav + (size_t)batch * NSAMP + (size_t)t0 * HOP;
        const int rem = NSAMP - t0 * HOP;   // valid samples in this segment
        for (int i = tid; i < SEG / 2; i += 256) {
            const int n0 = 2 * i;
            float x0 = (n0     < rem) ? wsrc[n0]     : 0.0f;
            float x1 = (n0 + 1 < rem) ? wsrc[n0 + 1] : 0.0f;
            v2h pk = __builtin_amdgcn_cvt_pkrtz(x0, x1);
            ((v2h*)sw)[i + (i >> 7)] = pk;  // dword pad every 128 dwords
        }
    }
    __syncthreads();

    const int t    = lane & 15;
    const int koff = (lane & 16) ? 8 : 0;

    // ---- Each wave: 16-bin x 64-frame job, 8 accumulators ------------------
    for (int job = wave; job < NBT; job += 8) {
        const int binbase = job * 16;

        // Opaque zero: stops the compiler from proving B loads job-invariant
        // and hoisting 64 B fragments into registers (>256 VGPRs, MSB storm).
        int jz = 0;
        asm volatile("" : "+v"(jz));

        const v16h* __restrict__ aRp =
            (const v16h*)a16 + ((size_t)job * NKC) * 32 + lane;
        const v16h* __restrict__ aIp =
            (const v16h*)a16 + ((size_t)(job + 32) * NKC) * 32 + lane;
        const float* rowR = kern + (size_t)(binbase + t) * KROW + koff;
        const float* rowI = rowR + (size_t)IMAG_OFF * KROW;

        v8f accR[NTT] = {{}, {}, {}, {}};
        v8f accI[NTT] = {{}, {}, {}, {}};

#pragma unroll 2
        for (int kki = 0; kki < NKC; ++kki) {
            v16h ar, ai;
            if (PRE) {
                ar = aRp[kki * 32];
                ai = aIp[kki * 32];
            } else {
                ar = cvt_a_frag(rowR + kki * 32);
                ai = cvt_a_frag(rowI + kki * 32);
            }
            const int kpad = jz + kki * 32 + ((kki >= 8) ? 2 : 0) + koff;
#pragma unroll
            for (int nt = 0; nt < NTT; ++nt) {
                const v16h b = load_b_raw(sw, (t + 16 * nt) * PADHOP + kpad);
                accR[nt] = __builtin_amdgcn_wmma_f32_16x16x32_f16(
                    false, ar, false, b, (short)0, accR[nt], false, false);
                accI[nt] = __builtin_amdgcn_wmma_f32_16x16x32_f16(
                    false, ai, false, b, (short)0, accI[nt], false, false);
            }
        }

        // ---- Epilogue: mag/pha, coalesced over frames ----------------------
        const int mrow = (lane & 16) ? 8 : 0;       // C/D: VGPR r -> M = r(+8)
#pragma unroll
        for (int nt = 0; nt < NTT; ++nt) {
            const int tcol = t0 + nt * 16 + t;
            if (tcol >= NFRM) continue;
#pragma unroll
            for (int r = 0; r < 8; ++r) {
                const int bin = binbase + mrow + r;
                if (bin < NBIN) {
                    const float re = accR[nt][r];
                    const float im = accI[nt][r];
                    const float m  = sqrtf(re * re + im * im + EPS_F);
                    const float p  = atan2f(im, re);
                    const size_t idx =
                        ((size_t)batch * NBIN + bin) * NFRM + tcol;
                    mag[idx] = m;
                    pha[idx] = p;
                }
            }
        }
    }
}

extern "C" void kernel_launch(void* const* d_in, const int* in_sizes, int n_in,
                              void* d_out, int out_size, void* d_ws, size_t ws_size,
                              hipStream_t stream) {
    (void)in_sizes; (void)n_in; (void)out_size;
    const float* wav  = (const float*)d_in[0];   // (16, 512000) f32
    const float* kern = (const float*)d_in[1];   // (1024, 1, 512) f32
    float* mag = (float*)d_out;                              // (16,257,1999)
    float* pha = mag + (size_t)NBATCH * NBIN * NFRM;         // (16,257,1999)

    const int nftiles = (NFRM + FT_PER_BLK - 1) / FT_PER_BLK;  // 32
    dim3 grid(nftiles, NBATCH);

    if (ws_size >= A16_BYTES) {
        _Float16* a16 = (_Float16*)d_ws;
        swizzle_a16_kernel<<<A16_GROUPS / 256, 256, 0, stream>>>(kern, a16);
        stft_wmma_kernel<true><<<grid, 256, 0, stream>>>(wav, kern, a16, mag, pha);
    } else {
        stft_wmma_kernel<false><<<grid, 256, 0, stream>>>(wav, kern, nullptr, mag, pha);
    }
}
